// Generator_59700045414727
// MI455X (gfx1250) — matrix operand order repaired and checked by reading, not compiled, not run on hardware
//
#include <hip/hip_runtime.h>
#include <hip/hip_bf16.h>
#include <stdint.h>

typedef _Float16 v16h __attribute__((ext_vector_type(16)));
typedef _Float16 v8h  __attribute__((ext_vector_type(8)));
typedef float    v8f  __attribute__((ext_vector_type(8)));

#define B_SZ 512
#define T_SZ 128
#define D_SZ 64
#define H_SZ 512
#define C_SZ 10
#define LD_A0 (D_SZ + H_SZ)   // 576
#define LD_A1 (2 * H_SZ)      // 1024
#define MISSING_V 128.0f

__device__ __forceinline__ float sigmoid_(float x) { return 1.0f / (1.0f + __expf(-x)); }
__device__ __forceinline__ float tanh_(float x)    { return 1.0f - 2.0f / (__expf(2.0f * x) + 1.0f); }

__device__ __forceinline__ v8f wmma_f16(v16h a, v16h b, v8f c) {
  // D = A(16x32 f16) * B(32x16 f16) + C(16x16 f32)
  return __builtin_amdgcn_wmma_f32_16x16x32_f16(false, a, false, b, (short)0, c, false, false);
}

// Assemble the per-lane 16-element A operand from two contiguous 8xf16 chunks.
// p must already point at this lane's first chunk (row m, K = kBase + 8*g);
// second chunk is at +16 halves (CDNA5 ISA 7.12.2 A 16x32 layout).
__device__ __forceinline__ v16h load_a_chunks(const _Float16* __restrict__ p) {
  v8h lo = *(const v8h*)(p);
  v8h hi = *(const v8h*)(p + 16);
  v16h a;
#pragma unroll
  for (int i = 0; i < 8; ++i) { a[i] = lo[i]; a[8 + i] = hi[i]; }
  return a;
}

// Pack row-major f32 weights [K x N] (rows 0..rows0-1 from M0, rest from M1)
// into WMMA B-operand tiles: tile (nT,kt) is 512 f16; lane l = khalf*16+n holds
// B[kt*32 + khalf*8 + (e&7) + (e>>3)*16][nT*16 + n] at element e (contiguous 32B per lane).
__global__ void pack_b_kernel(const float* __restrict__ M0, int rows0,
                              const float* __restrict__ M1, int ncols,
                              int numK, _Float16* __restrict__ out, int total) {
  for (int idx = blockIdx.x * blockDim.x + threadIdx.x; idx < total;
       idx += gridDim.x * blockDim.x) {
    int e    = idx & 15;
    int l    = (idx >> 4) & 31;
    int tile = idx >> 9;
    int kt   = tile % numK;
    int nT   = tile / numK;
    int khalf = l >> 4, n = l & 15;
    int k   = kt * 32 + khalf * 8 + (e & 7) + (e >> 3) * 16;
    int col = nT * 16 + n;
    float v = (k < rows0) ? M0[(size_t)k * ncols + col]
                          : M1[(size_t)(k - rows0) * ncols + col];
    out[idx] = (_Float16)v;
  }
}

// One fused LSTM layer step: z = Aact @ Wpk + bias -> gates -> c,h update.
// Each wave owns a 32(M)x16(H) h-tile: 2 A tiles x 4 gate B-tiles -> 8 v_wmma
// per K-tile. B pointers are four named streaming pointers (no pointer array)
// so address-space inference keeps them global (global_load_b128, LOADcnt only).
__global__ __launch_bounds__(128)
void lstm_layer_kernel(const _Float16* __restrict__ Aact, int K,
                       const _Float16* __restrict__ Wpk,
                       const float* __restrict__ bias,
                       float* __restrict__ cbuf,
                       _Float16* __restrict__ hOutA, int ldA, int colA,
                       _Float16* __restrict__ hOutB, int ldB, int colB,
                       float* __restrict__ hOutF) {
  const int lane  = threadIdx.x;                    // 0..31 (wave32)
  const int hTile = blockIdx.x * 4 + threadIdx.y;   // 0..31
  const int mPair = blockIdx.y;                     // 0..15 -> rows mPair*32..+31
  const int numK  = K >> 5;

  const int g = lane >> 4;      // K-half selector for A operand
  const int m = lane & 15;      // row within 16-row A tile

  // Per-lane streaming pointers (advance by constants each K-tile).
  const _Float16* a0p = Aact + (size_t)(mPair * 32 + m) * K + 8 * g;
  const _Float16* a1p = a0p + (size_t)16 * K;
  const size_t gateStride = (size_t)32 * numK * 512;   // 32 n-tiles per gate
  const _Float16* bp0 = Wpk + (size_t)hTile * numK * 512 + lane * 16;
  const _Float16* bp1 = bp0 + gateStride;
  const _Float16* bp2 = bp1 + gateStride;
  const _Float16* bp3 = bp2 + gateStride;

  v8f acc0[4] = {};   // rows mPair*32 .. +15
  v8f acc1[4] = {};   // rows mPair*32+16 .. +31
  for (int kt = 0; kt < numK; ++kt) {
    v16h a0 = load_a_chunks(a0p);
    v16h a1 = load_a_chunks(a1p);
    a0p += 32; a1p += 32;
    v16h bm0 = *(const v16h*)(bp0); bp0 += 512;
    v16h bm1 = *(const v16h*)(bp1); bp1 += 512;
    v16h bm2 = *(const v16h*)(bp2); bp2 += 512;
    v16h bm3 = *(const v16h*)(bp3); bp3 += 512;
    acc0[0] = wmma_f16(a0, bm0, acc0[0]);
    acc1[0] = wmma_f16(a1, bm0, acc1[0]);
    acc0[1] = wmma_f16(a0, bm1, acc0[1]);
    acc1[1] = wmma_f16(a1, bm1, acc1[1]);
    acc0[2] = wmma_f16(a0, bm2, acc0[2]);
    acc1[2] = wmma_f16(a1, bm2, acc1[2]);
    acc0[3] = wmma_f16(a0, bm3, acc0[3]);
    acc1[3] = wmma_f16(a1, bm3, acc1[3]);
  }

  // D layout: lane n = col within tile; VGPR r -> row = 8*(lane>=16)+r
  const int n    = lane & 15;
  const int half = lane >> 4;
  const int col  = hTile * 16 + n;
  const float bi = bias[0 * H_SZ + col];
  const float bf = bias[1 * H_SZ + col];
  const float bg = bias[2 * H_SZ + col];
  const float bo = bias[3 * H_SZ + col];
  const bool hasB = (hOutB != nullptr);
  const bool hasF = (hOutF != nullptr);
#pragma unroll
  for (int mi = 0; mi < 2; ++mi) {
    const v8f* acc = mi ? acc1 : acc0;
#pragma unroll
    for (int r = 0; r < 8; ++r) {
      int b = mPair * 32 + mi * 16 + half * 8 + r;
      size_t ci = (size_t)b * H_SZ + col;
      float cold = cbuf[ci];
      float cn = sigmoid_(acc[1][r] + bf) * cold +
                 sigmoid_(acc[0][r] + bi) * tanh_(acc[2][r] + bg);
      float hn = sigmoid_(acc[3][r] + bo) * tanh_(cn);
      cbuf[ci] = cn;
      hOutA[(size_t)b * ldA + colA + col] = (_Float16)hn;
      if (hasB) hOutB[(size_t)b * ldB + colB + col] = (_Float16)hn;
      if (hasF) hOutF[ci] = hn;
    }
  }
}

// pred = h1_prev @ W + b; write prediction row (t-1); impute cur_t -> A0.cur
__global__ __launch_bounds__(128)
void pred_impute_kernel(const _Float16* __restrict__ h1p,
                        const _Float16* __restrict__ Wpk,   // K=512,N=64 packed
                        const float* __restrict__ bvec,
                        const float* __restrict__ x, int t,
                        float* __restrict__ predOut,
                        _Float16* __restrict__ curOut) {
  const int lane  = threadIdx.x;
  const int nTile = threadIdx.y;   // 0..3 (D=64)
  const int mTile = blockIdx.x;    // 0..31
  const int g = lane >> 4, m = lane & 15;
  const _Float16* ap = h1p + (size_t)(mTile * 16 + m) * H_SZ + 8 * g;
  const _Float16* bp = Wpk + ((size_t)(nTile * 32)) * 512 + lane * 16;
  v8f acc = {};
  for (int kt = 0; kt < 32; ++kt) {
    v16h a  = load_a_chunks(ap);
    v16h bm = *(const v16h*)(bp);
    ap += 32; bp += 512;
    acc = wmma_f16(a, bm, acc);
  }
  const int n = lane & 15, half = lane >> 4;
  const int d = nTile * 16 + n;
  const float bb = bvec[d];
#pragma unroll
  for (int r = 0; r < 8; ++r) {
    int b = mTile * 16 + half * 8 + r;
    float p  = acc[r] + bb;
    float xv = x[(size_t)b * T_SZ * D_SZ + (size_t)t * D_SZ + d];
    float cur = (xv == MISSING_V) ? p : xv;
    predOut[(size_t)b * ((T_SZ - 1) * D_SZ) + (size_t)(t - 1) * D_SZ + d] = p;
    curOut[(size_t)b * LD_A0 + d] = (_Float16)cur;
  }
}

__global__ void zerof_kernel(float* p, int n) {
  for (int i = blockIdx.x * blockDim.x + threadIdx.x; i < n; i += gridDim.x * blockDim.x)
    p[i] = 0.0f;
}
__global__ void zeroh_kernel(_Float16* p, int n) {
  for (int i = blockIdx.x * blockDim.x + threadIdx.x; i < n; i += gridDim.x * blockDim.x)
    p[i] = (_Float16)0.0f;
}
__global__ void copyf_kernel(const float* __restrict__ s, float* __restrict__ d, int n) {
  for (int i = blockIdx.x * blockDim.x + threadIdx.x; i < n; i += gridDim.x * blockDim.x)
    d[i] = s[i];
}
// step 0 input: raw x (no imputation), f16 into A0[0].cur
__global__ void x0_to_cur_kernel(const float* __restrict__ x, _Float16* __restrict__ A0) {
  for (int i = blockIdx.x * blockDim.x + threadIdx.x; i < B_SZ * D_SZ;
       i += gridDim.x * blockDim.x) {
    int b = i / D_SZ, d = i % D_SZ;
    A0[(size_t)b * LD_A0 + d] = (_Float16)x[(size_t)b * T_SZ * D_SZ + d];
  }
}

// sum of ((tgt - pred) * mask)^2 -> accum (block reduce + atomic)
__global__ void predloss_kernel(const float* __restrict__ pred,
                                const float* __restrict__ tgt,
                                const float* __restrict__ msk,
                                float* __restrict__ accum, int n) {
  __shared__ float red[256];
  float s = 0.0f;
  for (int i = blockIdx.x * blockDim.x + threadIdx.x; i < n; i += gridDim.x * blockDim.x) {
    float dlt = (tgt[i] - pred[i]) * msk[i];
    s += dlt * dlt;
  }
  red[threadIdx.x] = s;
  __syncthreads();
  for (int off = 128; off > 0; off >>= 1) {
    if ((int)threadIdx.x < off) red[threadIdx.x] += red[threadIdx.x + off];
    __syncthreads();
  }
  if (threadIdx.x == 0) atomicAdd(accum, red[0]);
}

// logits, log-softmax class loss, accuracy; combine with prediction loss
__global__ void final_kernel(const float* __restrict__ h1f,
                             const float* __restrict__ Wc,
                             const float* __restrict__ bc,
                             const float* __restrict__ label,
                             const float* __restrict__ accum,
                             float* __restrict__ lossOut,
                             float* __restrict__ accOut) {
  int b = blockIdx.x * blockDim.x + threadIdx.x;
  if (b >= B_SZ) return;
  float logits[C_SZ];
#pragma unroll
  for (int c = 0; c < C_SZ; ++c) logits[c] = bc[c];
  for (int k = 0; k < H_SZ; ++k) {
    float hv = h1f[(size_t)b * H_SZ + k];
#pragma unroll
    for (int c = 0; c < C_SZ; ++c) logits[c] += hv * Wc[(size_t)k * C_SZ + c];
  }
  float mx = logits[0]; int ai = 0;
  for (int c = 1; c < C_SZ; ++c) if (logits[c] > mx) { mx = logits[c]; ai = c; }
  float se = 0.0f;
  for (int c = 0; c < C_SZ; ++c) se += __expf(logits[c] - mx);
  float lse = mx + __logf(se);
  float dot = 0.0f, lmax = -1.0f; int li = 0;
  for (int c = 0; c < C_SZ; ++c) {
    float lv = label[(size_t)b * C_SZ + c];
    dot += lv * (logits[c] - lse);
    if (lv > lmax) { lmax = lv; li = c; }
  }
  float lossPred = accum[0] / ((float)B_SZ * (T_SZ - 1) * D_SZ) / (float)B_SZ;
  lossOut[b] = -dot + 1.0f * lossPred;   // LAMDA = 1
  accOut[b]  = (ai == li) ? 1.0f : 0.0f;
}

extern "C" void kernel_launch(void* const* d_in, const int* in_sizes, int n_in,
                              void* d_out, int out_size, void* d_ws, size_t ws_size,
                              hipStream_t stream) {
  const float* x     = (const float*)d_in[0];
  const float* ptgt  = (const float*)d_in[1];
  const float* msk   = (const float*)d_in[2];
  const float* label = (const float*)d_in[3];
  const float* W     = (const float*)d_in[4];
  const float* bvec  = (const float*)d_in[5];
  const float* k0    = (const float*)d_in[6];
  const float* r0    = (const float*)d_in[7];
  const float* b0    = (const float*)d_in[8];
  const float* k1    = (const float*)d_in[9];
  const float* r1    = (const float*)d_in[10];
  const float* b1    = (const float*)d_in[11];
  const float* Wc    = (const float*)d_in[12];
  const float* bc    = (const float*)d_in[13];

  const int nPred = B_SZ * (T_SZ - 1) * D_SZ;
  float* outLoss = (float*)d_out;
  float* outPred = outLoss + B_SZ;
  float* outMask = outPred + (size_t)nPred;
  float* outAcc  = outMask + (size_t)nPred;

  // workspace carve-up (256B aligned)
  char* wp = (char*)d_ws;
  auto carve = [&](size_t bytes) -> char* {
    char* p = wp;
    wp += (bytes + 255) & ~(size_t)255;
    return p;
  };
  _Float16* W0p   = (_Float16*)carve((size_t)(D_SZ + H_SZ) * 4 * H_SZ * 2); // 576x2048
  _Float16* W1p   = (_Float16*)carve((size_t)(2 * H_SZ) * 4 * H_SZ * 2);    // 1024x2048
  _Float16* Wpp   = (_Float16*)carve((size_t)H_SZ * D_SZ * 2);              // 512x64
  _Float16* A0b[2], *A1b[2];
  A0b[0] = (_Float16*)carve((size_t)B_SZ * LD_A0 * 2);
  A0b[1] = (_Float16*)carve((size_t)B_SZ * LD_A0 * 2);
  A1b[0] = (_Float16*)carve((size_t)B_SZ * LD_A1 * 2);
  A1b[1] = (_Float16*)carve((size_t)B_SZ * LD_A1 * 2);
  _Float16* h1p = (_Float16*)carve((size_t)B_SZ * H_SZ * 2);
  float* h1f   = (float*)carve((size_t)B_SZ * H_SZ * 4);
  float* c0    = (float*)carve((size_t)B_SZ * H_SZ * 4);
  float* c1    = (float*)carve((size_t)B_SZ * H_SZ * 4);
  float* accum = (float*)carve(256);

  // 1) pack weights into WMMA B-operand layout (f16), every call (deterministic)
  pack_b_kernel<<<4096, 256, 0, stream>>>(k0, D_SZ, r0, 4 * H_SZ, (D_SZ + H_SZ) / 32,
                                          W0p, (D_SZ + H_SZ) * 4 * H_SZ);
  pack_b_kernel<<<4096, 256, 0, stream>>>(k1, H_SZ, r1, 4 * H_SZ, (2 * H_SZ) / 32,
                                          W1p, 2 * H_SZ * 4 * H_SZ);
  pack_b_kernel<<<256, 256, 0, stream>>>(W, H_SZ, W, D_SZ, H_SZ / 32, Wpp, H_SZ * D_SZ);

  // 2) init state
  zerof_kernel<<<512, 256, 0, stream>>>(c0, B_SZ * H_SZ);
  zerof_kernel<<<512, 256, 0, stream>>>(c1, B_SZ * H_SZ);
  zerof_kernel<<<1, 32, 0, stream>>>(accum, 1);
  zeroh_kernel<<<512, 256, 0, stream>>>(A0b[0], B_SZ * LD_A0);
  zeroh_kernel<<<512, 256, 0, stream>>>(A1b[0], B_SZ * LD_A1);
  x0_to_cur_kernel<<<128, 256, 0, stream>>>(x, A0b[0]);

  dim3 lblock(32, 4), lgrid(8, 16);     // 32 hTiles x 16 mPairs, 4 waves/block
  dim3 pblock(32, 4), pgrid(32);        // 4 nTiles x 32 mTiles

  // 3) step 0 (raw input, zero state)
  lstm_layer_kernel<<<lgrid, lblock, 0, stream>>>(A0b[0], LD_A0, W0p, b0, c0,
                                                  A0b[1], LD_A0, D_SZ,
                                                  A1b[0], LD_A1, 0, nullptr);
  lstm_layer_kernel<<<lgrid, lblock, 0, stream>>>(A1b[0], LD_A1, W1p, b1, c1,
                                                  A1b[1], LD_A1, H_SZ,
                                                  h1p, H_SZ, 0, h1f);
  // 4) steps 1..T-1 (ping-pong activation buffers)
  for (int t = 1; t < T_SZ; ++t) {
    int cur = t & 1, nxt = (t + 1) & 1;
    pred_impute_kernel<<<pgrid, pblock, 0, stream>>>(h1p, Wpp, bvec, x, t,
                                                     outPred, A0b[cur]);
    lstm_layer_kernel<<<lgrid, lblock, 0, stream>>>(A0b[cur], LD_A0, W0p, b0, c0,
                                                    A0b[nxt], LD_A0, D_SZ,
                                                    A1b[cur], LD_A1, 0, nullptr);
    lstm_layer_kernel<<<lgrid, lblock, 0, stream>>>(A1b[cur], LD_A1, W1p, b1, c1,
                                                    A1b[nxt], LD_A1, H_SZ,
                                                    h1p, H_SZ, 0, h1f);
  }

  // 5) tail: mask passthrough, masked-MSE reduction, class loss + accuracy
  copyf_kernel<<<1024, 256, 0, stream>>>(msk, outMask, nPred);
  predloss_kernel<<<1024, 256, 0, stream>>>(outPred, ptgt, msk, accum, nPred);
  final_kernel<<<2, 256, 0, stream>>>(h1f, Wc, bc, label, accum, outLoss, outAcc);
}
